// T5Attention_68375879352597
// MI455X (gfx1250) — matrix operand order repaired and checked by reading, not compiled, not run on hardware
//
#include <hip/hip_runtime.h>

// ---------------------------------------------------------------------------
// T5 attention for MI455X (gfx1250), wave32 + V_WMMA_F32_16X16X32_BF16.
//   cvt:    query f32 -> bf16 once (A-fragments become pure b128 loads)
//   prep:   weights -> bf16 transposed, staged via GLOBAL_LOAD_ASYNC_TO_LDS
//   proj:   fused QKV GEMM (all-bf16 fragments), 64x16 register block,
//           scale folded into Q, V stored transposed per (b,h)
//   flash:  32 queries/wave, online softmax, 16 WMMA per 32-key chunk,
//           output stored directly as bf16
//   out:    O-projection GEMM (bf16 A), f32 result
//   bias:   bucket table (16x4095) + 268MB float4 tensor writer
// ---------------------------------------------------------------------------

typedef __attribute__((ext_vector_type(16))) __bf16 v16bf;
typedef __attribute__((ext_vector_type(8)))  __bf16 v8bf;
typedef __attribute__((ext_vector_type(8)))  float  v8f;
typedef __attribute__((address_space(3)))    float  lds_f32;

#define D_MODEL 1024
#define SEQ     2048
#define BATCH   2
#define HEADS   16
#define HDIM    64
#define TABW    4095          // relative distances -2047..2047

__device__ __forceinline__ v8f wmma_bf16(v16bf a, v16bf b, v8f c) {
  return __builtin_amdgcn_wmma_f32_16x16x32_bf16(
      /*neg_a=*/false, a, /*neg_b=*/false, b,
      /*c_mod=*/(short)0, c, /*reuse_a=*/false, /*reuse_b=*/false);
}

// ---- 64x16 C block: C = A(bf16,row-major)[M,K] x W (via W^T bf16 [N,K]) ----
// A frag (ISA 7.12.2): row = m + (lane&15); K runs [s..s+7],[16+s..16+s+7].
// B frag: col = n0 + (lane&15); K runs [s..s+7],[16+s..16+s+7] like A.
__device__ __forceinline__ void gemm_block64(
    const __bf16* __restrict__ A, int lda, const __bf16* __restrict__ BT, int ldb,
    int m0, int n0, int K, int lane, v8f acc[4]) {
  const int s    = (lane & 16) ? 8 : 0;
  const int koff = (lane & 16) ? 16 : 0;
  const __bf16* arow0 = A  + (size_t)(m0 + (lane & 15)) * lda;
  const __bf16* brow  = BT + (size_t)(n0 + (lane & 15)) * ldb;
  for (int kk = 0; kk < K; kk += 32) {
    v16bf b;
#pragma unroll
    for (int e = 0; e < 16; ++e) b[e] = brow[kk + s + (e & 7) + ((e & 8) << 1)];
#pragma unroll
    for (int t = 0; t < 4; ++t) {
      const __bf16* arow = arow0 + (size_t)t * 16 * lda;
      v16bf a;
#pragma unroll
      for (int i = 0; i < 8; ++i) {
        a[i]     = arow[kk + s + i];
        a[8 + i] = arow[kk + 16 + s + i];
      }
      acc[t] = wmma_bf16(a, b, acc[t]);
    }
  }
}

// ---------------- query f32 -> bf16, vectorized ------------------------------
__global__ void __launch_bounds__(256)
k_cvt_bf16(const float* __restrict__ src, __bf16* __restrict__ dst, int n8) {
  int i = blockIdx.x * blockDim.x + threadIdx.x;
  if (i >= n8) return;
  const float4* s4 = (const float4*)src + (size_t)i * 2;
  float4 a = s4[0], b = s4[1];
  v8bf o;
  o[0] = (__bf16)a.x; o[1] = (__bf16)a.y; o[2] = (__bf16)a.z; o[3] = (__bf16)a.w;
  o[4] = (__bf16)b.x; o[5] = (__bf16)b.y; o[6] = (__bf16)b.z; o[7] = (__bf16)b.w;
  *((v8bf*)dst + i) = o;
}

// ---------------- weight transpose + bf16 convert: dst[n][k] = src[k][n] ----
// Global -> LDS staging uses the gfx1250 async copy engine (ASYNCcnt).
__global__ void __launch_bounds__(256)
k_w_t(const float* __restrict__ src, __bf16* __restrict__ dst, int K, int N) {
  __shared__ float tile[32][33];
  int n0 = blockIdx.x * 32, k0 = blockIdx.y * 32;
  int tx = threadIdx.x, ty = threadIdx.y;        // 32 x 8
#pragma unroll
  for (int r = 0; r < 32; r += 8) {
    unsigned lds_off = (unsigned)(size_t)(lds_f32*)&tile[ty + r][tx];
    unsigned long long gaddr =
        (unsigned long long)(size_t)(src + (size_t)(k0 + ty + r) * N + n0 + tx);
    asm volatile("global_load_async_to_lds_b32 %0, %1, off"
                 :: "v"(lds_off), "v"(gaddr) : "memory");
  }
  asm volatile("s_wait_asynccnt 0x0" ::: "memory");
  __syncthreads();
#pragma unroll
  for (int r = 0; r < 32; r += 8)
    dst[(size_t)(n0 + ty + r) * K + k0 + tx] = (__bf16)tile[tx][ty + r];
}

// ---------------- bias bucket table: tab[h][d+2047], d = j - i ---------------
__global__ void k_bias_table(const float* __restrict__ relb, float* __restrict__ tab) {
  int idx = blockIdx.x * blockDim.x + threadIdx.x;
  if (idx >= HEADS * TABW) return;
  int h = idx / TABW;
  int rp = (idx % TABW) - 2047;      // rp = j - i
  int ret = -rp;                     // i - j
  int sign = (ret < 0) ? 1 : 0;
  int a = (ret < 0) ? -ret : ret;
  int bucket;
  if (a < 8) {
    bucket = a;
  } else {
    float lf = logf((float)a / 8.0f + 1e-6f) * (8.0f / logf(16.0f));
    int li = (int)lf;
    bucket = 8 + (li > 7 ? 7 : li);
  }
  bucket += sign * 16;
  tab[idx] = relb[h * 32 + bucket];
}

// ---------------- fused QKV projection -------------------------------------
__global__ void __launch_bounds__(256)
k_proj_qkv(const __bf16* __restrict__ qcvt, const __bf16* __restrict__ WqT,
           const __bf16* __restrict__ WkvT, __bf16* __restrict__ qb,
           __bf16* __restrict__ kb, __bf16* __restrict__ vT) {
  int lane = threadIdx.x & 31;
  int wave = (blockIdx.x * blockDim.x + threadIdx.x) >> 5;
  const int NT = (3 * D_MODEL) / 16;             // 192
  int nt = wave % NT, mt = wave / NT;            // mt in [0,64)
  int m0 = mt * 64, n0 = nt * 16;

  const __bf16* BT; int nb;
  if (n0 < D_MODEL) { BT = WqT;  nb = n0; }
  else              { BT = WkvT; nb = n0 - D_MODEL; }

  v8f acc[4];
#pragma unroll
  for (int t = 0; t < 4; ++t) acc[t] = (v8f){0.f,0.f,0.f,0.f,0.f,0.f,0.f,0.f};
  gemm_block64(qcvt, D_MODEL, BT, D_MODEL, m0, nb, D_MODEL, lane, acc);

  int hh = (lane & 16) ? 8 : 0;
  int lc = lane & 15;
  if (n0 < D_MODEL) {                            // Q: row-major, scale folded
    int col = n0 + lc;
#pragma unroll
    for (int t = 0; t < 4; ++t)
#pragma unroll
      for (int r = 0; r < 8; ++r)
        qb[(size_t)(m0 + t*16 + r + hh) * D_MODEL + col] = (__bf16)(acc[t][r] * 0.125f);
  } else if (n0 < 2 * D_MODEL) {                 // K: row-major
    int col = n0 - D_MODEL + lc;
#pragma unroll
    for (int t = 0; t < 4; ++t)
#pragma unroll
      for (int r = 0; r < 8; ++r)
        kb[(size_t)(m0 + t*16 + r + hh) * D_MODEL + col] = (__bf16)acc[t][r];
  } else {                                       // V: transposed vT[b*1024+col][tok]
    int col = n0 - 2 * D_MODEL + lc;
#pragma unroll
    for (int t = 0; t < 4; ++t)
#pragma unroll
      for (int r = 0; r < 8; ++r) {
        int m = m0 + t*16 + r + hh;
        int bb = m >> 11, tok = m & (SEQ - 1);
        vT[(size_t)(bb * D_MODEL + col) * SEQ + tok] = (__bf16)acc[t][r];
      }
  }
}

// ---------------- flash attention: one wave per (b, h, 32-query block) -------
__global__ void __launch_bounds__(256)
k_flash_attn(const __bf16* __restrict__ qb, const __bf16* __restrict__ kb,
             const __bf16* __restrict__ vT, const float* __restrict__ tab,
             __bf16* __restrict__ attnbf) {
  __shared__ __bf16 plds[8][2 * 16 * 32];        // wave-private P slabs

  int lane   = threadIdx.x & 31;
  int wavein = threadIdx.x >> 5;
  int wave   = blockIdx.x * 8 + wavein;          // 2048 waves total
  int qt = wave & 63;                            // 64 query blocks of 32
  int h  = (wave >> 6) & 15;
  int b  = wave >> 10;
  int q0 = qt * 32;

  const __bf16* qbase = qb + (size_t)b * SEQ * D_MODEL + h * HDIM;
  const __bf16* kbase = kb + (size_t)b * SEQ * D_MODEL + h * HDIM;
  const __bf16* vbase = vT + (size_t)(b * HEADS + h) * HDIM * SEQ;  // [d][tok]
  const float*  tabh  = tab + h * TABW + 2047;

  const int mcol = lane & 15;
  const int s    = (lane & 16) ? 8 : 0;
  const int koff = (lane & 16) ? 16 : 0;
  const int hh   = s;

  // Q A-fragments: [q-tile u][k-chunk c]
  v16bf aq[2][2];
#pragma unroll
  for (int u = 0; u < 2; ++u) {
    const __bf16* qrow = qbase + (size_t)(q0 + u*16 + mcol) * D_MODEL;
#pragma unroll
    for (int i = 0; i < 8; ++i) {
      aq[u][0][i]     = qrow[s + i];       aq[u][0][8 + i] = qrow[16 + s + i];
      aq[u][1][i]     = qrow[32 + s + i];  aq[u][1][8 + i] = qrow[48 + s + i];
    }
  }

  float mrow[2][8], lrow[2][8];
  v8f acc[2][4];
#pragma unroll
  for (int u = 0; u < 2; ++u) {
#pragma unroll
    for (int r = 0; r < 8; ++r) { mrow[u][r] = -3.0e38f; lrow[u][r] = 0.0f; }
#pragma unroll
    for (int t = 0; t < 4; ++t) acc[u][t] = (v8f){0.f,0.f,0.f,0.f,0.f,0.f,0.f,0.f};
  }

  __bf16* pl = &plds[wavein][0];

  for (int kt = 0; kt < SEQ; kt += 32) {
    if (kt + 32 < SEQ) {
      __builtin_prefetch(kbase + (size_t)(kt + 32 + mcol) * D_MODEL, 0, 1);
      __builtin_prefetch(vbase + (size_t)(mcol * 4) * SEQ + kt + 32, 0, 1);
    }
    // ---- K B-fragments for this 32-key chunk: bk[j (16-key tile)][k-chunk] --
    v16bf bk[2][2];
#pragma unroll
    for (int j = 0; j < 2; ++j) {
      const __bf16* krow = kbase + (size_t)(kt + j*16 + mcol) * D_MODEL;
#pragma unroll
      for (int e = 0; e < 8; ++e) {
        bk[j][0][e] = krow[s + e];            bk[j][0][8 + e] = krow[16 + s + e];
        bk[j][1][e] = krow[32 + s + e];       bk[j][1][8 + e] = krow[48 + s + e];
      }
    }
    // ---- scores + online softmax per q-tile --------------------------------
#pragma unroll
    for (int u = 0; u < 2; ++u) {
      v8f st[2];
#pragma unroll
      for (int j = 0; j < 2; ++j) {
        v8f c = {0.f,0.f,0.f,0.f,0.f,0.f,0.f,0.f};
        c = wmma_bf16(aq[u][0], bk[j][0], c);
        c = wmma_bf16(aq[u][1], bk[j][1], c);
#pragma unroll
        for (int r = 0; r < 8; ++r)
          c[r] += tabh[(kt + j*16 + mcol) - (q0 + u*16 + r + hh)];
        st[j] = c;
      }
#pragma unroll
      for (int r = 0; r < 8; ++r) {
        float v = fmaxf(st[0][r], st[1][r]);
#pragma unroll
        for (int m = 8; m >= 1; m >>= 1) v = fmaxf(v, __shfl_xor(v, m, 32));
        float mn = fmaxf(mrow[u][r], v);
        float rs = __expf(mrow[u][r] - mn);
        mrow[u][r] = mn;
        lrow[u][r] *= rs;
#pragma unroll
        for (int t = 0; t < 4; ++t) acc[u][t][r] *= rs;
        float p0 = __expf(st[0][r] - mn);
        float p1 = __expf(st[1][r] - mn);
        float ps = p0 + p1;
#pragma unroll
        for (int m = 8; m >= 1; m >>= 1) ps += __shfl_xor(ps, m, 32);
        lrow[u][r] += ps;
        pl[u*512 + (r + hh)*32 + mcol]      = (__bf16)p0;
        pl[u*512 + (r + hh)*32 + 16 + mcol] = (__bf16)p1;
      }
    }
    asm volatile("s_wait_dscnt 0" ::: "memory");   // per-wave DS stores visible
    // ---- P back in A-fragment layout ---------------------------------------
    v16bf pa[2];
#pragma unroll
    for (int u = 0; u < 2; ++u) {
      const __bf16* prow = pl + u*512 + mcol * 32;
#pragma unroll
      for (int i = 0; i < 8; ++i) { pa[u][i] = prow[koff + i]; pa[u][8 + i] = prow[koff + 8 + i]; }
    }
    // ---- O += P . V : V^T fragments are contiguous along tokens ------------
#pragma unroll
    for (int t = 0; t < 4; ++t) {
      const __bf16* vrow = vbase + (size_t)(t*16 + mcol) * SEQ + kt + koff;
      v16bf bv;
#pragma unroll
      for (int e = 0; e < 16; ++e) bv[e] = vrow[e];
#pragma unroll
      for (int u = 0; u < 2; ++u) acc[u][t] = wmma_bf16(pa[u], bv, acc[u][t]);
    }
  }
  // ---- normalize and store bf16 [B,L,D] ------------------------------------
  __bf16* obase = attnbf + (size_t)b * SEQ * D_MODEL + h * HDIM;
#pragma unroll
  for (int u = 0; u < 2; ++u)
#pragma unroll
    for (int r = 0; r < 8; ++r) {
      float inv = 1.0f / lrow[u][r];
      int qi = q0 + u*16 + r + hh;
#pragma unroll
      for (int t = 0; t < 4; ++t)
        obase[(size_t)qi * D_MODEL + t*16 + mcol] = (__bf16)(acc[u][t][r] * inv);
    }
}

// ---------------- output projection: out = attn @ Wo -------------------------
__global__ void __launch_bounds__(256)
k_out_proj(const __bf16* __restrict__ attnbf, const __bf16* __restrict__ WoT,
           float* __restrict__ out) {
  int lane = threadIdx.x & 31;
  int wave = (blockIdx.x * blockDim.x + threadIdx.x) >> 5;
  const int NT = D_MODEL / 16;                   // 64
  int nt = wave % NT, mt = wave / NT;            // mt in [0,64)
  int m0 = mt * 64, n0 = nt * 16;
  v8f acc[4];
#pragma unroll
  for (int t = 0; t < 4; ++t) acc[t] = (v8f){0.f,0.f,0.f,0.f,0.f,0.f,0.f,0.f};
  gemm_block64(attnbf, D_MODEL, WoT, D_MODEL, m0, n0, D_MODEL, lane, acc);
  int hh = (lane & 16) ? 8 : 0;
  int col = n0 + (lane & 15);
#pragma unroll
  for (int t = 0; t < 4; ++t)
#pragma unroll
    for (int r = 0; r < 8; ++r)
      out[(size_t)(m0 + t*16 + r + hh) * D_MODEL + col] = acc[t][r];
}

// ---------------- 268 MB bias tensor writer (float4, grid-stride) ------------
__global__ void k_bias_write(const float* __restrict__ tab, float* __restrict__ bout) {
  size_t stride = (size_t)gridDim.x * blockDim.x;
  const size_t total4 = (size_t)HEADS * SEQ * SEQ / 4;
  for (size_t v = blockIdx.x * blockDim.x + threadIdx.x; v < total4; v += stride) {
    size_t idx = v * 4;
    int j = (int)(idx & 2047);
    int i = (int)((idx >> 11) & 2047);
    int h = (int)(idx >> 22);
    const float* t = tab + h * TABW + (j - i) + 2047;
    float4 o = make_float4(t[0], t[1], t[2], t[3]);
    ((float4*)bout)[v] = o;
  }
}

// ---------------------------------------------------------------------------
extern "C" void kernel_launch(void* const* d_in, const int* in_sizes, int n_in,
                              void* d_out, int out_size, void* d_ws, size_t ws_size,
                              hipStream_t stream) {
  (void)in_sizes; (void)n_in; (void)out_size; (void)ws_size;
  const float* query = (const float*)d_in[0];   // [2,2048,1024]
  const float* Wq    = (const float*)d_in[1];   // [1024,1024]
  const float* Wkv   = (const float*)d_in[2];   // [1024,2048]
  const float* Wo    = (const float*)d_in[3];   // [1024,1024]
  const float* relb  = (const float*)d_in[4];   // [512,1]

  const size_t TOK = (size_t)BATCH * SEQ;       // 4096
  __bf16* qbuf   = (__bf16*)d_ws;               //  8 MiB  Q (scaled, bf16)
  __bf16* kbuf   = qbuf + TOK * D_MODEL;        //  8 MiB  K
  __bf16* vT     = kbuf + TOK * D_MODEL;        //  8 MiB  V^T [b*1024+col][tok]
  __bf16* attnbf = vT + TOK * D_MODEL;          //  8 MiB  attention output (bf16)
  __bf16* qcvt   = attnbf + TOK * D_MODEL;      //  8 MiB  query bf16
  __bf16* WqT    = qcvt + TOK * D_MODEL;        //  2 MiB
  __bf16* WkvT   = WqT + (size_t)D_MODEL * D_MODEL;      // 4 MiB
  __bf16* WoT    = WkvT + (size_t)2 * D_MODEL * D_MODEL; // 2 MiB
  float*  tab    = (float*)(WoT + (size_t)D_MODEL * D_MODEL); // 256 KiB

  float* out     = (float*)d_out;               // [2,2048,1024]
  float* biasout = out + TOK * D_MODEL;         // [1,16,2048,2048]

  k_cvt_bf16<<<2048, 256, 0, stream>>>(query, qcvt, (int)(TOK * D_MODEL / 8));
  k_w_t<<<dim3(32, 32), dim3(32, 8), 0, stream>>>(Wq,  WqT,  D_MODEL, D_MODEL);
  k_w_t<<<dim3(64, 32), dim3(32, 8), 0, stream>>>(Wkv, WkvT, D_MODEL, 2*D_MODEL);
  k_w_t<<<dim3(32, 32), dim3(32, 8), 0, stream>>>(Wo,  WoT,  D_MODEL, D_MODEL);
  k_bias_table<<<(HEADS * TABW + 255) / 256, 256, 0, stream>>>(relb, tab);
  k_proj_qkv <<<1536, 256, 0, stream>>>(qcvt, WqT, WkvT, qbuf, kbuf, vT);
  k_flash_attn<<<256, 256, 0, stream>>>(qbuf, kbuf, vT, tab, attnbf);
  k_out_proj <<<512, 256, 0, stream>>>(attnbf, WoT, out);
  k_bias_write<<<4096, 256, 0, stream>>>(tab, biasout);
}
